// GCN_19396072308969
// MI455X (gfx1250) — matrix operand order repaired and checked by reading, not compiled, not run on hardware
//
#include <hip/hip_runtime.h>
#include <hip/hip_bf16.h>

// ---------------------------------------------------------------------------
// GCN forward for MI455X (gfx1250, wave32, WMMA).
//   h1 = scatter_add(ew * (x@W1)[src] -> dst) + b1
//   h2 = scatter_add(ew * (h1@W2)[src] -> dst) + b2
//   out = softmax((h2@lw1 + lb1) @ lw2 + lb2)    [head fused per node]
//
// GEMMs: v_wmma_f32_16x16x32_bf16.  Activations pre-converted to bf16;
// weights pre-packed into exact WMMA B-fragment order so each lane's
// fragment is one contiguous 32-byte load (2x global_load_b128).
// ---------------------------------------------------------------------------

typedef __attribute__((ext_vector_type(16))) __bf16 v16bf;
typedef __attribute__((ext_vector_type(8)))  __bf16 v8bf;
typedef __attribute__((ext_vector_type(4)))  __bf16 v4bf;
typedef __attribute__((ext_vector_type(8)))  float  v8f;

#define GCN_H   128
#define GCN_HID 64
#define GCN_OUT 40
#define KSTEPS  (GCN_H / 32)       // 4
#define NTILES  (GCN_H / 16)       // 8
#define PACKED_B_ELEMS (NTILES * KSTEPS * 32 * 16)   // 16384 bf16 = 32 KB

// ---- elementwise fp32 -> bf16 (4 values / thread) -------------------------
__global__ void __launch_bounds__(256)
f32_to_bf16(const float* __restrict__ in, __bf16* __restrict__ out, long total4) {
    long i = (long)blockIdx.x * blockDim.x + threadIdx.x;
    if (i >= total4) return;
    float4 v = ((const float4*)in)[i];
    v4bf o = { (__bf16)v.x, (__bf16)v.y, (__bf16)v.z, (__bf16)v.w };
    ((v4bf*)out)[i] = o;
}

// ---- pack W (128x128 f32 row-major) into WMMA B-fragment order ------------
// B fragment layout (ISA 7.12.2, 16-bit B 32x16): lane = 16*half + col,
// element i (0..15) holds K = ks*32 + 8*half + i (i<8) / ks*32 + 16 + 8*half + (i-8) (i>=8), column = nt*16 + col.
__global__ void __launch_bounds__(256)
pack_b_frag(const float* __restrict__ W, __bf16* __restrict__ P) {
    int t = blockIdx.x * blockDim.x + threadIdx.x;    // 16384 threads
    if (t >= PACKED_B_ELEMS) return;
    int i    = t & 15;
    int lane = (t >> 4) & 31;
    int ks   = (t >> 9) & (KSTEPS - 1);
    int nt   = t >> 11;
    int half = lane >> 4;
    int col  = lane & 15;
    int k = ks * 32 + (i < 8 ? 8 * half + i : 16 + 8 * half + (i - 8));
    int n = nt * 16 + col;
    P[t] = (__bf16)W[k * GCN_H + n];
}

// ---- C(MxH) = A(MxH, bf16 row-major) @ Bpacked, fp32 out ------------------
// One wave -> one 16x16 tile; 8 waves/block -> one 16-row strip of all 8
// column tiles.  A fragment: lane = row + 16*half; two contiguous 8-elem
// K-chunks at k0+8*half and k0+16+8*half (one b128 load each).
__global__ void __launch_bounds__(256)
gemm_wmma_bf16(const __bf16* __restrict__ A, const __bf16* __restrict__ Bp,
               float* __restrict__ C, int M) {
    const int wave = threadIdx.x >> 5;     // = n-tile index (NTILES == 8 waves)
    const int lane = threadIdx.x & 31;
    const int m0   = blockIdx.x * 16;
    if (m0 >= M) return;

    const int half = lane >> 4;
    const int rc   = lane & 15;            // row (A) / col (B,C)

    const __bf16* arow = A + (long)(m0 + rc) * GCN_H + 8 * half;
    const __bf16* bptr = Bp + ((long)(wave * KSTEPS) * 32 + lane) * 16;

    __builtin_prefetch(bptr, 0, 3);        // global_prefetch_b8: warm weights

    v8f acc = {};
#pragma unroll
    for (int ks = 0; ks < KSTEPS; ++ks) {
        union { v16bf v; v8bf h[2]; } a;
        a.h[0] = *(const v8bf*)(arow + ks * 32);        // K = k0+8h .. +8
        a.h[1] = *(const v8bf*)(arow + ks * 32 + 16);   // K = k0+16+8h .. +8
        v16bf b = *(const v16bf*)(bptr + ks * 32 * 16); // contiguous 32 B
        acc = __builtin_amdgcn_wmma_f32_16x16x32_bf16(
            /*neg_a=*/false, a.v, /*neg_b=*/false, b,
            /*c_mod=*/(short)0, acc, /*reuse_a=*/false, /*reuse_b=*/false);
    }

    // C/D layout: VGPR v -> row m0 + v + 8*half, col = wave*16 + rc
#pragma unroll
    for (int v = 0; v < 8; ++v)
        C[(long)(m0 + v + 8 * half) * GCN_H + wave * 16 + rc] = acc[v];
}

// ---- out[node][f] = bias[f] (aggregation accumulator init) ----------------
__global__ void __launch_bounds__(256)
init_bias(float* __restrict__ out, const float* __restrict__ b, long total) {
    long i = (long)blockIdx.x * blockDim.x + threadIdx.x;
    if (i < total) out[i] = b[i & (GCN_H - 1)];
}

// ---- weighted gather + atomic scatter-add, float4 per thread --------------
__global__ void __launch_bounds__(256)
scatter_edges(const float* __restrict__ hw, const int* __restrict__ src,
              const int* __restrict__ dst, const float* __restrict__ ew,
              float* __restrict__ out, long total) {
    long t = (long)blockIdx.x * blockDim.x + threadIdx.x;
    if (t >= total) return;
    int e = (int)(t >> 5);                 // 32 float4-chunks per edge
    int c = (int)(t & 31);
    int s = src[e], d = dst[e];
    float w = ew[e];
    float4 v = ((const float4*)(hw + (long)s * GCN_H))[c];
    float* o = out + (long)d * GCN_H + c * 4;
    atomicAdd(o + 0, v.x * w);
    atomicAdd(o + 1, v.y * w);
    atomicAdd(o + 2, v.z * w);
    atomicAdd(o + 3, v.w * w);
}

// ---- fused MLP head + softmax: one wave per node --------------------------
__global__ void __launch_bounds__(256)
head_kernel(const float* __restrict__ h, const float* __restrict__ lw1,
            const float* __restrict__ lb1, const float* __restrict__ lw2,
            const float* __restrict__ lb2, float* __restrict__ out) {
    __shared__ float smem[8 * 256];
    const int wave = threadIdx.x >> 5;
    const int lane = threadIdx.x & 31;
    float* ws = smem + wave * 256;         // [0,128) h, [128,192) t, [192,232) u
    const int node = blockIdx.x * 8 + wave;

    const float* hr = h + (long)node * GCN_H;
#pragma unroll
    for (int i = 0; i < 4; ++i) ws[lane + 32 * i] = hr[lane + 32 * i];
    __syncthreads();

#pragma unroll
    for (int rep = 0; rep < 2; ++rep) {
        int col = lane + 32 * rep;
        float acc = lb1[col];
        for (int k = 0; k < GCN_H; ++k) acc = fmaf(ws[k], lw1[k * GCN_HID + col], acc);
        ws[128 + col] = acc;
    }
    __syncthreads();

    for (int col = lane; col < GCN_OUT; col += 32) {
        float acc = lb2[col];
        for (int k = 0; k < GCN_HID; ++k) acc = fmaf(ws[128 + k], lw2[k * GCN_OUT + col], acc);
        ws[192 + col] = acc;
    }
    __syncthreads();

    float m = -3.4e38f;
    for (int j = 0; j < GCN_OUT; ++j) m = fmaxf(m, ws[192 + j]);
    float s = 0.f;
    for (int j = 0; j < GCN_OUT; ++j) s += __expf(ws[192 + j] - m);
    float inv = 1.f / s;
    for (int col = lane; col < GCN_OUT; col += 32)
        out[(long)node * GCN_OUT + col] = __expf(ws[192 + col] - m) * inv;
}

extern "C" void kernel_launch(void* const* d_in, const int* in_sizes, int n_in,
                              void* d_out, int out_size, void* d_ws, size_t ws_size,
                              hipStream_t stream) {
    const float* x   = (const float*)d_in[0];
    const int*   ei  = (const int*)d_in[1];
    const float* ew  = (const float*)d_in[2];
    const float* W1  = (const float*)d_in[3];
    const float* b1  = (const float*)d_in[4];
    const float* W2  = (const float*)d_in[5];
    const float* b2  = (const float*)d_in[6];
    const float* lw1 = (const float*)d_in[7];
    const float* lb1 = (const float*)d_in[8];
    const float* lw2 = (const float*)d_in[9];
    const float* lb2 = (const float*)d_in[10];

    const int N = in_sizes[0] / GCN_H;   // 50000
    const int E = in_sizes[2];           // 800000
    const int* src = ei;
    const int* dst = ei + E;

    const long nh = (long)N * GCN_H;

    // workspace carve-up (all chunks 32-byte aligned by construction)
    float*  buf0 = (float*)d_ws;                     // GEMM output     (N*H f32)
    float*  buf1 = buf0 + nh;                        // agg accumulator (N*H f32)
    __bf16* abf  = (__bf16*)(buf1 + nh);             // bf16 activations(N*H)
    __bf16* pW1  = abf + nh;                         // packed W1 (32 KB)
    __bf16* pW2  = pW1 + PACKED_B_ELEMS;             // packed W2 (32 KB)

    const long edgework  = (long)E * (GCN_H / 4);
    const int  gemm_grid = N / 16;                           // 3125
    const int  conv_grid = (int)((nh / 4 + 255) / 256);
    const int  init_grid = (int)((nh + 255) / 256);
    const int  scat_grid = (int)((edgework + 255) / 256);
    const int  pack_grid = (PACKED_B_ELEMS + 255) / 256;     // 64

    // ---- one-time weight packing + input conversion ----
    pack_b_frag<<<pack_grid, 256, 0, stream>>>(W1, pW1);
    pack_b_frag<<<pack_grid, 256, 0, stream>>>(W2, pW2);
    f32_to_bf16<<<conv_grid, 256, 0, stream>>>(x, abf, nh / 4);

    // ---- layer 1 ----
    gemm_wmma_bf16<<<gemm_grid, 256, 0, stream>>>(abf, pW1, buf0, N);
    init_bias<<<init_grid, 256, 0, stream>>>(buf1, b1, nh);
    scatter_edges<<<scat_grid, 256, 0, stream>>>(buf0, src, dst, ew, buf1, edgework);

    // ---- layer 2 ----
    f32_to_bf16<<<conv_grid, 256, 0, stream>>>(buf1, abf, nh / 4);
    gemm_wmma_bf16<<<gemm_grid, 256, 0, stream>>>(abf, pW2, buf0, N);
    init_bias<<<init_grid, 256, 0, stream>>>(buf1, b2, nh);
    scatter_edges<<<scat_grid, 256, 0, stream>>>(buf0, src, dst, ew, buf1, edgework);

    // ---- fused MLP head + softmax ----
    head_kernel<<<N / 8, 256, 0, stream>>>(buf1, lw1, lb1, lw2, lb2, (float*)d_out);
}
